// MultiHeadAttention_26104811225412
// MI455X (gfx1250) — matrix operand order repaired and checked by reading, not compiled, not run on hardware
//
#include <hip/hip_runtime.h>
#include <hip/hip_bf16.h>
#include <stdint.h>

// Problem constants (fixed by reference)
#define BB 8
#define SS 1024
#define DD 1024
#define HH 16
#define DHD 64
// M = BB*SS = 8192

typedef __attribute__((ext_vector_type(16))) __bf16        bf16x16;
typedef __attribute__((ext_vector_type(8)))  float         f32x8;
typedef __attribute__((ext_vector_type(4)))  float         f32x4;
typedef __attribute__((ext_vector_type(4)))  unsigned int  u32x4;

union FragAB {
    bf16x16  v;
    uint16_t u[16];
    u32x4    q[2];
};

__device__ inline uint16_t f2bf(float f) {
    union { float f; uint32_t u; } x; x.f = f;
    uint32_t r = x.u + 0x7FFFu + ((x.u >> 16) & 1u);  // round-to-nearest-even
    return (uint16_t)(r >> 16);
}

__device__ inline f32x8 wmma_bf16(const FragAB& a, const FragAB& b, f32x8 c) {
    return __builtin_amdgcn_wmma_f32_16x16x32_bf16(false, a.v, false, b.v,
                                                   (short)0, c, false, false);
}

// ---------------------------------------------------------------------------
// Tiled GEMM + bias: out = A[8192,1024] @ W[1024,1024] + bias
//   A_BF16: A is bf16 (uint16 storage) else f32 (converted while staging LDS)
//   OUT_MODE 0: f32 row-major [M, D]
//   OUT_MODE 1: bf16 head layout   [b, h, s, dh]
//   OUT_MODE 2: bf16 head layout T [b, h, dh, s]   (for V)
// Block: 256 threads (8 waves), tile 128x128, BK=32. Wave = 32x64 (2x4 tiles).
// Double-buffered LDS: 1 barrier / K-step, next tile's global loads overlap
// the current tile's WMMAs.
// ---------------------------------------------------------------------------
template <bool A_BF16, int OUT_MODE>
__global__ __launch_bounds__(256) void gemm_bias(const void* __restrict__ Ap,
                                                 const float* __restrict__ W,
                                                 const float* __restrict__ bias,
                                                 void* __restrict__ outp) {
    __shared__ uint16_t lA[2][128][40];   // [buf][m][k] (+8 pad -> conflict-free)
    __shared__ uint16_t lBt[2][128][40];  // [buf][n][k] transposed W tile

    const int tid  = threadIdx.x;
    const int lane = tid & 31;
    const int wid  = tid >> 5;   // 0..7
    const int wm   = wid >> 1;   // 0..3 : 32-row slice
    const int wn   = wid & 1;    // 0..1 : 64-col slice
    const int lr   = lane & 15;
    const int lh   = lane >> 4;
    const int bn   = blockIdx.x * 128;
    const int bm   = blockIdx.y * 128;

    const int arow = tid >> 1, aseg = (tid & 1) * 16;   // A tile: 16 contig elems
    const int brow = tid >> 3, bseg = (tid & 7) * 16;   // W tile: 16 contig elems

    f32x8 acc[2][4];
#pragma unroll
    for (int i = 0; i < 2; i++)
#pragma unroll
        for (int j = 0; j < 4; j++)
#pragma unroll
            for (int r = 0; r < 8; r++) acc[i][j][r] = 0.0f;

    // staging registers for the next tile
    f32x4 ga[4];
    u32x4 ga16[2];
    f32x4 gb[4];

    auto issue_loads = [&](int kt) {
        if (A_BF16) {
            const u32x4* g = (const u32x4*)((const uint16_t*)Ap +
                             (size_t)(bm + arow) * DD + kt + aseg);
            ga16[0] = g[0]; ga16[1] = g[1];
        } else {
            const f32x4* g = (const f32x4*)((const float*)Ap +
                             (size_t)(bm + arow) * DD + kt + aseg);
            ga[0] = g[0]; ga[1] = g[1]; ga[2] = g[2]; ga[3] = g[3];
        }
        const f32x4* g = (const f32x4*)(W + (size_t)(kt + brow) * DD + bn + bseg);
        gb[0] = g[0]; gb[1] = g[1]; gb[2] = g[2]; gb[3] = g[3];
    };
    auto store_tiles = [&](int buf) {
        if (A_BF16) {
            *(u32x4*)&lA[buf][arow][aseg]     = ga16[0];
            *(u32x4*)&lA[buf][arow][aseg + 8] = ga16[1];
        } else {
            float fv[16];
            *(f32x4*)&fv[0] = ga[0]; *(f32x4*)&fv[4]  = ga[1];
            *(f32x4*)&fv[8] = ga[2]; *(f32x4*)&fv[12] = ga[3];
            uint16_t t[16];
#pragma unroll
            for (int i = 0; i < 16; i++) t[i] = f2bf(fv[i]);
            *(u32x4*)&lA[buf][arow][aseg]     = *(u32x4*)&t[0];
            *(u32x4*)&lA[buf][arow][aseg + 8] = *(u32x4*)&t[8];
        }
        float fv[16];
        *(f32x4*)&fv[0] = gb[0]; *(f32x4*)&fv[4]  = gb[1];
        *(f32x4*)&fv[8] = gb[2]; *(f32x4*)&fv[12] = gb[3];
#pragma unroll
        for (int i = 0; i < 16; i++) lBt[buf][bseg + i][brow] = f2bf(fv[i]);
    };

    issue_loads(0);
    store_tiles(0);

#pragma unroll 2
    for (int kt = 0; kt < DD; kt += 32) {
        const int cur = (kt >> 5) & 1;
        const int nxt = cur ^ 1;
        const bool more = (kt + 32) < DD;
        if (more) issue_loads(kt + 32);   // overlap with this step's WMMAs
        __syncthreads();                  // lA/lBt[cur] visible; [nxt] reads done

        FragAB a[2], b[4];
#pragma unroll
        for (int mt = 0; mt < 2; mt++) {
            int row = wm * 32 + mt * 16 + lr;
            int k0  = lh * 8;
            a[mt].q[0] = *(const u32x4*)&lA[cur][row][k0];
            a[mt].q[1] = *(const u32x4*)&lA[cur][row][k0 + 16];
        }
#pragma unroll
        for (int nt = 0; nt < 4; nt++) {
            int col = wn * 64 + nt * 16 + lr;
            int k0  = lh * 8;
            b[nt].q[0] = *(const u32x4*)&lBt[cur][col][k0];
            b[nt].q[1] = *(const u32x4*)&lBt[cur][col][k0 + 16];
        }
#pragma unroll
        for (int mt = 0; mt < 2; mt++)
#pragma unroll
            for (int nt = 0; nt < 4; nt++)
                acc[mt][nt] = wmma_bf16(a[mt], b[nt], acc[mt][nt]);

        if (more) store_tiles(nxt);
    }

    // Epilogue: C layout = col lr, rows r + 8*lh
#pragma unroll
    for (int mt = 0; mt < 2; mt++) {
#pragma unroll
        for (int nt = 0; nt < 4; nt++) {
            int   n  = bn + wn * 64 + nt * 16 + lr;
            float bv = bias[n];
#pragma unroll
            for (int r = 0; r < 8; r++) {
                int   m   = bm + wm * 32 + mt * 16 + r + lh * 8;
                float val = acc[mt][nt][r] + bv;
                if (OUT_MODE == 0) {
                    ((float*)outp)[(size_t)m * DD + n] = val;
                } else {
                    int b_ = m >> 10, s_ = m & 1023;
                    int h_ = n >> 6,  d_ = n & 63;
                    size_t idx = (OUT_MODE == 1)
                        ? ((((size_t)b_ * HH + h_) * SS + s_) * DHD + d_)
                        : ((((size_t)b_ * HH + h_) * DHD + d_) * SS + s_);
                    ((uint16_t*)outp)[idx] = f2bf(val);
                }
            }
        }
    }
}

// ---------------------------------------------------------------------------
// Flash attention: each wave independently owns 16 query rows of one (b,h).
// Q[b,h,s,dh] bf16, K[b,h,s,dh] bf16, VT[b,h,dh,s] bf16 -> out bf16 [b,s,h*dh]
// ---------------------------------------------------------------------------
__global__ __launch_bounds__(256) void attn_kernel(const uint16_t* __restrict__ Q,
                                                   const uint16_t* __restrict__ Kk,
                                                   const uint16_t* __restrict__ VT,
                                                   const unsigned char* __restrict__ mask,
                                                   uint16_t* __restrict__ outA) {
    __shared__ uint16_t lP[8][16][136];  // per-wave P tile (16 q x 128 k), padded

    const int tid = threadIdx.x, lane = tid & 31, wid = tid >> 5;
    const int lr = lane & 15, lh = lane >> 4;

    int task = blockIdx.x * 8 + wid;     // B*H*(S/16) = 8192 wave tasks
    int b    = task >> 10;               // 1024 tasks per batch
    int rem  = task & 1023;
    int h    = rem >> 6;
    int q0   = (rem & 63) << 4;
    const size_t bh = (size_t)(b * HH + h);

    // Q A-fragments (K=64 -> 2 chunks), resident in registers
    FragAB qf[2];
    {
        const uint16_t* qrow = Q + (bh * SS + q0 + lr) * DHD;
        int k0 = lh * 8;
        qf[0].q[0] = *(const u32x4*)(qrow + k0);
        qf[0].q[1] = *(const u32x4*)(qrow + k0 + 16);
        qf[1].q[0] = *(const u32x4*)(qrow + 32 + k0);
        qf[1].q[1] = *(const u32x4*)(qrow + 32 + k0 + 16);
    }
    float mq[8];
#pragma unroll
    for (int r = 0; r < 8; r++)
        mq[r] = mask[b * SS + q0 + r + lh * 8] ? 1.0f : 0.0f;

    float run_max[8], run_sum[8];
    f32x8 o[4];
#pragma unroll
    for (int r = 0; r < 8; r++) { run_max[r] = -3.0e38f; run_sum[r] = 0.0f; }
#pragma unroll
    for (int t = 0; t < 4; t++)
#pragma unroll
        for (int r = 0; r < 8; r++) o[t][r] = 0.0f;

    for (int kb = 0; kb < SS; kb += 128) {
        f32x8 sc[8];
#pragma unroll
        for (int nt = 0; nt < 8; nt++)
#pragma unroll
            for (int r = 0; r < 8; r++) sc[nt][r] = 0.0f;

        // scores: Q @ K^T  (B-frag: col = key row, contiguous dh run)
#pragma unroll
        for (int nt = 0; nt < 8; nt++) {
            const uint16_t* krow = Kk + (bh * SS + kb + nt * 16 + lr) * DHD;
            int k0 = lh * 8;
            FragAB b0, b1;
            b0.q[0] = *(const u32x4*)(krow + k0);
            b0.q[1] = *(const u32x4*)(krow + k0 + 16);
            b1.q[0] = *(const u32x4*)(krow + 32 + k0);
            b1.q[1] = *(const u32x4*)(krow + 32 + k0 + 16);
            sc[nt] = wmma_bf16(qf[0], b0, sc[nt]);
            sc[nt] = wmma_bf16(qf[1], b1, sc[nt]);
        }

        // mask + scale + per-lane partial row max
        float tmax[8];
#pragma unroll
        for (int r = 0; r < 8; r++) tmax[r] = -3.0e38f;
#pragma unroll
        for (int nt = 0; nt < 8; nt++) {
            float mk = mask[b * SS + kb + nt * 16 + lr] ? 1.0f : 0.0f;
#pragma unroll
            for (int r = 0; r < 8; r++) {
                float v = sc[nt][r] * 0.125f;            // 1/sqrt(64)
                if (mk + mq[r] > 0.0f) v = -1.0e9f;
                sc[nt][r] = v;
                tmax[r]   = fmaxf(tmax[r], v);
            }
        }
        // cross-lane row max within the 16-lane C-layout group
#pragma unroll
        for (int r = 0; r < 8; r++) {
            float v = tmax[r];
            v = fmaxf(v, __shfl_xor(v, 1));
            v = fmaxf(v, __shfl_xor(v, 2));
            v = fmaxf(v, __shfl_xor(v, 4));
            v = fmaxf(v, __shfl_xor(v, 8));
            tmax[r] = v;
        }

        float corr[8], rs[8];
#pragma unroll
        for (int r = 0; r < 8; r++) {
            float nm   = fmaxf(run_max[r], tmax[r]);
            corr[r]    = __expf(run_max[r] - nm);
            run_max[r] = nm;
            rs[r]      = 0.0f;
        }

        // P = exp(s - m), write bf16 to per-wave LDS (C-layout -> A-layout xpose)
#pragma unroll
        for (int nt = 0; nt < 8; nt++) {
#pragma unroll
            for (int r = 0; r < 8; r++) {
                float p = __expf(sc[nt][r] - run_max[r]);
                rs[r] += p;
                lP[wid][r + lh * 8][nt * 16 + lr] = f2bf(p);
            }
        }
#pragma unroll
        for (int r = 0; r < 8; r++) {
            float v = rs[r];
            v += __shfl_xor(v, 1);
            v += __shfl_xor(v, 2);
            v += __shfl_xor(v, 4);
            v += __shfl_xor(v, 8);
            run_sum[r] = run_sum[r] * corr[r] + v;
        }
#pragma unroll
        for (int t = 0; t < 4; t++)
#pragma unroll
            for (int r = 0; r < 8; r++) o[t][r] *= corr[r];

        // O += P @ V   (4 K-chunks x 4 dh tiles)
#pragma unroll
        for (int c = 0; c < 4; c++) {
            FragAB pf;
            const uint16_t* prow = &lP[wid][lr][c * 32 + lh * 8];
            pf.q[0] = *(const u32x4*)(prow);
            pf.q[1] = *(const u32x4*)(prow + 16);
#pragma unroll
            for (int t = 0; t < 4; t++) {
                FragAB vf;
                const uint16_t* vrow = VT + (bh * DHD + t * 16 + lr) * SS +
                                       kb + c * 32 + lh * 8;
                vf.q[0] = *(const u32x4*)(vrow);
                vf.q[1] = *(const u32x4*)(vrow + 16);
                o[t] = wmma_bf16(pf, vf, o[t]);
            }
        }
    }

    // normalize + store bf16 [b, s, h*dh]
#pragma unroll
    for (int t = 0; t < 4; t++) {
#pragma unroll
        for (int r = 0; r < 8; r++) {
            float inv = __builtin_amdgcn_rcpf(run_sum[r]);
            float val = o[t][r] * inv;
            int   qi  = q0 + r + lh * 8;
            int   ft  = h * DHD + t * 16 + lr;
            outA[((size_t)(b * SS + qi)) * DD + ft] = f2bf(val);
        }
    }
}

// ---------------------------------------------------------------------------
extern "C" void kernel_launch(void* const* d_in, const int* in_sizes, int n_in,
                              void* d_out, int out_size, void* d_ws, size_t ws_size,
                              hipStream_t stream) {
    const float*         x    = (const float*)d_in[0];
    const unsigned char* mask = (const unsigned char*)d_in[1];  // jax bool = 1 byte
    const float* WQ = (const float*)d_in[2];
    const float* bQ = (const float*)d_in[3];
    const float* WK = (const float*)d_in[4];
    const float* bK = (const float*)d_in[5];
    const float* WV = (const float*)d_in[6];
    const float* bV = (const float*)d_in[7];
    const float* WO = (const float*)d_in[8];
    const float* bO = (const float*)d_in[9];

    char* ws = (char*)d_ws;                       // 64 MB used
    uint16_t* Qb  = (uint16_t*)(ws);                              // 16 MB
    uint16_t* Kb  = (uint16_t*)(ws + (size_t)16 * 1024 * 1024);   // 16 MB
    uint16_t* VTb = (uint16_t*)(ws + (size_t)32 * 1024 * 1024);   // 16 MB
    uint16_t* Ab  = (uint16_t*)(ws + (size_t)48 * 1024 * 1024);   // 16 MB

    dim3 grid(8, 64), blk(256);
    gemm_bias<false, 1><<<grid, blk, 0, stream>>>(x, WQ, bQ, Qb);
    gemm_bias<false, 1><<<grid, blk, 0, stream>>>(x, WK, bK, Kb);
    gemm_bias<false, 2><<<grid, blk, 0, stream>>>(x, WV, bV, VTb);
    attn_kernel<<<1024, blk, 0, stream>>>(Qb, Kb, VTb, mask, Ab);
    gemm_bias<true, 0><<<grid, blk, 0, stream>>>(Ab, WO, bO, (float*)d_out);
}
